// GNN_graphpred_23665269801392
// MI455X (gfx1250) — compile-verified, matrix operand order checked by reading, not run on hardware
//
#include <hip/hip_runtime.h>

#define BN_INV 0.9999950000374997f   // 1/sqrt(1+1e-5), matches reference _bn_* scaling

typedef __attribute__((ext_vector_type(16))) _Float16 v16h;
typedef __attribute__((ext_vector_type(8)))  _Float16 h8v;   // 16-byte LDS/global vector
typedef __attribute__((ext_vector_type(8)))  float    v8f;

__device__ __forceinline__ float elu_f(float v) { return v > 0.f ? v : __expf(v) - 1.f; }

// Assemble a 16-half WMMA operand from two 16B vectors.
// Documented 16-bit operand layout (ISA 7.12.2): lanes 0-15 hold K={0..7,16..23},
// lanes 16-31 hold K={8..15,24..31} -> slots (hi) and (2+hi) of a 4-slot (32-half) K run.
__device__ __forceinline__ v16h frag16(const _Float16* rowbase, int hi) {
  const h8v* p = (const h8v*)rowbase;
  h8v lo = p[hi];
  h8v hv = p[2 + hi];
  v16h r;
  #pragma unroll
  for (int j = 0; j < 8; ++j) { r[j] = lo[j]; r[j + 8] = hv[j]; }
  return r;
}

// ---------------- small utility kernels ----------------

__global__ void k_cvt_f16(const float* __restrict__ s, _Float16* __restrict__ d, int n) {
  int i = blockIdx.x * 256 + threadIdx.x;
  if (i < n) d[i] = (_Float16)s[i];
}

// conv weights -> (t, OC, IC) f16 ; swapflip=1 implements _tconv_w (flip taps, swap O/I)
__global__ void k_convw(const float* __restrict__ s, _Float16* __restrict__ d,
                        int OC, int IC, int swapflip) {
  int i = blockIdx.x * 256 + threadIdx.x;
  int n = 15 * OC * IC;
  if (i >= n) return;
  int t = i / (OC * IC); int r = i % (OC * IC); int o = r / IC; int c = r % IC;
  float v = swapflip ? s[(c * OC + o) * 15 + (14 - t)]
                     : s[(o * IC + c) * 15 + t];
  d[i] = (_Float16)v;
}

__global__ void k_node_embed(const int* __restrict__ x,
    const float* __restrict__ xe1, const float* __restrict__ xe2, const float* __restrict__ xe3,
    const float* __restrict__ xe4, const float* __restrict__ xe5, const float* __restrict__ xe6,
    const float* __restrict__ xe7, float* __restrict__ h) {
  int i = blockIdx.x * 256 + threadIdx.x;  // 100000*256 elems
  int n = i >> 8, d = i & 255;
  const int* xr = x + n * 7;
  h[i] = xe1[xr[0]*256+d] + xe2[xr[1]*256+d] + xe3[xr[2]*256+d] + xe4[xr[3]*256+d]
       + xe5[xr[4]*256+d] + xe6[xr[5]*256+d] + xe7[xr[6]*256+d];
}

__global__ void k_init_agg(const float* __restrict__ h, float* __restrict__ agg,
    const float* __restrict__ e1, const float* __restrict__ e2,
    const float* __restrict__ e3, const float* __restrict__ e4) {
  int i = blockIdx.x * 256 + threadIdx.x;
  int d = i & 255;
  float sl = e1[4*256+d] + e2[d] + e3[d] + e4[d];   // self-loop embedding
  agg[i] = h[i] + sl;
}

__global__ void k_edge_scatter(const int* __restrict__ ei, const int* __restrict__ ea,
    const float* __restrict__ h, float* __restrict__ agg,
    const float* __restrict__ e1, const float* __restrict__ e2,
    const float* __restrict__ e3, const float* __restrict__ e4) {
  int e = blockIdx.x;        // one edge per block, one feature per lane
  int d = threadIdx.x;
  int s = ei[e], t = ei[400000 + e];
  const int* a = ea + e * 4;
  float v = h[(size_t)s*256 + d] + e1[a[0]*256+d] + e2[a[1]*256+d]
          + e3[a[2]*256+d] + e4[a[3]*256+d];
  atomicAdd(&agg[(size_t)t*256 + d], v);   // global_atomic_add_f32
}

// ---------------- WMMA GEMM: D = epi(A @ B + bias) ----------------
// A: (M,K) f32 or f16 ; B: (K,N) f16 ; block tile 128x64, 8 waves (wave32), each wave 32x32.
// LDS: As row-major (K contiguous), Bst transposed (N-major, K contiguous) so every
// fragment load is a 16B ds_load_b128. flags: 1=relu 2=elu 4=bn(g,bb) 8=f16 out
template<bool AF16>
__global__ void k_gemm(const void* __restrict__ Ap, const _Float16* __restrict__ B,
                       void* __restrict__ Dp, const float* __restrict__ bias,
                       const float* __restrict__ g, const float* __restrict__ bb,
                       int M, int N, int K, int flags) {
  __shared__ __align__(16) _Float16 As[128][40];   // 80B row stride, slots 16B aligned
  __shared__ __align__(16) _Float16 Bst[64][40];   // [n][k]
  const int tid = threadIdx.x;
  const int lane = tid & 31, wid = tid >> 5;
  const int wm = wid & 3, wn = wid >> 2;
  const int bm = blockIdx.y * 128, bn = blockIdx.x * 64;
  const int lh = lane & 15, hi = lane >> 4;
  v8f acc[2][2] = {};
  const int nk = K >> 5;
  for (int kc = 0; kc < nk; ++kc) {
    // ---- stage A (128 x 32), wide loads, convert to f16 ----
    if (AF16) {
      #pragma unroll
      for (int i = 0; i < 2; ++i) {               // 512 h8v's, 2 per thread
        int e  = i * 256 + tid;
        int r  = e >> 2, c8 = e & 3;
        int gm = bm + r;
        h8v v = {};
        if (gm < M)
          v = *(const h8v*)((const _Float16*)Ap + (size_t)gm * K + kc * 32 + c8 * 8);
        ((h8v*)&As[r][0])[c8] = v;
      }
    } else {
      #pragma unroll
      for (int i = 0; i < 4; ++i) {               // 1024 float4's, 4 per thread
        int e  = i * 256 + tid;
        int r  = e >> 3, c4 = e & 7;
        int gm = bm + r;
        float4 v = make_float4(0.f, 0.f, 0.f, 0.f);
        if (gm < M)
          v = *(const float4*)((const float*)Ap + (size_t)gm * K + kc * 32 + c4 * 4);
        As[r][c4*4 + 0] = (_Float16)v.x;
        As[r][c4*4 + 1] = (_Float16)v.y;
        As[r][c4*4 + 2] = (_Float16)v.z;
        As[r][c4*4 + 3] = (_Float16)v.w;
      }
    }
    // ---- stage B (32 x 64) transposed into Bst[n][k], one 16B load per thread ----
    {
      int r  = tid >> 3;        // k row 0..31
      int c8 = tid & 7;         // n group
      h8v v = *(const h8v*)(B + (size_t)(kc * 32 + r) * N + bn + c8 * 8);
      #pragma unroll
      for (int j = 0; j < 8; ++j) Bst[c8*8 + j][r] = v[j];
    }
    if (kc + 1 < nk) {                            // global_prefetch_b8 of next A chunk
      int gm = bm + (tid >> 1);
      if (gm < M) {
        if (AF16) __builtin_prefetch(&((const _Float16*)Ap)[(size_t)gm * K + (kc+1)*32], 0, 0);
        else      __builtin_prefetch(&((const float*)Ap)[(size_t)gm * K + (kc+1)*32], 0, 0);
      }
    }
    __syncthreads();
    v16h af[2], bf[2];
    #pragma unroll
    for (int s = 0; s < 2; ++s) af[s] = frag16(&As[wm*32 + s*16 + lh][0], hi);
    #pragma unroll
    for (int s = 0; s < 2; ++s) bf[s] = frag16(&Bst[wn*32 + s*16 + lh][0], hi);
    #pragma unroll
    for (int i = 0; i < 2; ++i)
      #pragma unroll
      for (int jj = 0; jj < 2; ++jj)
        acc[i][jj] = __builtin_amdgcn_wmma_f32_16x16x32_f16(
            false, af[i], false, bf[jj], (short)0, acc[i][jj], false, false);
    __syncthreads();
  }
  #pragma unroll
  for (int i = 0; i < 2; ++i)
    #pragma unroll
    for (int jj = 0; jj < 2; ++jj)
      #pragma unroll
      for (int r = 0; r < 8; ++r) {
        int gm = bm + wm*32 + i*16 + hi*8 + r;   // documented f32 C/D layout
        int gn = bn + wn*32 + jj*16 + lh;
        if (gm < M) {
          float v = acc[i][jj][r] + bias[gn];
          if (flags & 4) v = v * (g[gn] * BN_INV) + bb[gn];
          if (flags & 1) v = fmaxf(v, 0.f);
          if (flags & 2) v = elu_f(v);
          if (flags & 8) ((_Float16*)Dp)[(size_t)gm * N + gn] = (_Float16)v;
          else           ((float*)Dp)[(size_t)gm * N + gn] = v;
        }
      }
}

// ---------------- WMMA conv1d (NCH, OIH weights as (t,OC,IC) f16, pad 7, k=15) ----------------
// grid (L/64, OC/16, B), block 128 (4 waves), wave w owns 16 output columns.
// Input slab stored transposed (L-major, IC contiguous) so B fragments are b128 LDS loads.
__global__ void k_conv_wmma(const _Float16* __restrict__ X, const _Float16* __restrict__ WT,
                            _Float16* __restrict__ Y, const float* __restrict__ bias,
                            const float* __restrict__ g, const float* __restrict__ bb,
                            int IC, int OC, int flags) {
  __shared__ __align__(16) _Float16 Xst[78][40];   // [l][ic-chunk]
  const int tid = threadIdx.x;
  const int lane = tid & 31, w = tid >> 5;
  const int lh = lane & 15, hi = lane >> 4;
  const int l0 = blockIdx.x * 64;
  const int ot = blockIdx.y;
  const int b  = blockIdx.z;
  const _Float16* Xb = X + (size_t)b * IC * 512;
  v8f acc = {};
  for (int icc = 0; icc < (IC >> 5); ++icc) {
    for (int e = tid; e < 32 * 78; e += 128) {     // slab: 78 cols (halo 7+7) x 32 ch
      int r = e / 78, c = e % 78;                  // r = channel, c = column
      int gl = l0 - 7 + c;
      Xst[c][r] = (gl >= 0 && gl < 512) ? Xb[(size_t)(icc*32 + r)*512 + gl] : (_Float16)0.f;
    }
    __syncthreads();
    for (int t = 0; t < 15; ++t) {
      // A: weights, two 16B global loads (L2-resident)
      const _Float16* wrow = WT + ((size_t)(t * OC) + ot*16 + lh) * IC + icc*32;
      v16h af = frag16(wrow, hi);
      // B: input slab, two 16B LDS loads
      v16h bf = frag16(&Xst[w*16 + lh + t][0], hi);
      acc = __builtin_amdgcn_wmma_f32_16x16x32_f16(false, af, false, bf, (short)0, acc, false, false);
    }
    __syncthreads();
  }
  #pragma unroll
  for (int r = 0; r < 8; ++r) {
    int o = ot*16 + hi*8 + r;
    int l = l0 + w*16 + lh;
    float v = acc[r] + bias[o];
    if (flags & 4) v = v * (g[o] * BN_INV) + bb[o];
    if (flags & 1) v = fmaxf(v, 0.f);
    Y[(size_t)b * OC * 512 + (size_t)o * 512 + l] = (_Float16)v;
  }
}

// ---------------- misc data-movement ----------------

__global__ void k_hp_gather(const int* __restrict__ words, const _Float16* __restrict__ wemb,
                            _Float16* __restrict__ hp) {
  long long i = (long long)blockIdx.x * 256 + threadIdx.x;  // (B, D, L)
  int l = (int)(i & 511);
  int d = (int)((i >> 9) & 255);
  int b = (int)(i >> 17);
  hp[i] = wemb[(size_t)words[b*512 + l] * 256 + d];
}

__global__ void k_super(const float* __restrict__ h, const int* __restrict__ ptr,
                        float* __restrict__ sr) {
  int b = blockIdx.x, d = threadIdx.x;
  int n = ptr[b + 1] - 1;
  sr[b*256 + d] = h[(size_t)n*256 + d];
}

__global__ void k_a2t(const _Float16* __restrict__ A, _Float16* __restrict__ T) {
  long long i = (long long)blockIdx.x * 256 + threadIdx.x;  // (B*L, C=128)
  int c = (int)(i & 127);
  int l = (int)((i >> 7) & 511);
  int b = (int)(i >> 16);
  T[i] = A[(size_t)b*128*512 + (size_t)c*512 + l];
}

// ---------------- affine gate (_aff) ----------------
// pass A: masked mean-pool + 'g' branch -> xg (B,C)
__global__ void k_affA(const _Float16* __restrict__ X, const _Float16* __restrict__ R,
                       const float* __restrict__ mask,
                       const float* gw1, const float* gb1, const float* gg1, const float* gbb1,
                       const float* gw2, const float* gb2, const float* gg2, const float* gbb2,
                       float* __restrict__ xg, int C) {
  __shared__ float red[128];
  __shared__ float mp[128];
  __shared__ float h1s[16];
  __shared__ float msum_s;
  int b = blockIdx.x, t = threadIdx.x;
  const float* mrow = mask + b*512;
  float s = 0.f;
  for (int l = t; l < 512; l += 128) s += mrow[l];
  red[t] = s; __syncthreads();
  for (int st = 64; st > 0; st >>= 1) { if (t < st) red[t] += red[t+st]; __syncthreads(); }
  if (t == 0) msum_s = fmaxf(red[0], 1e-9f);
  __syncthreads();
  if (t < C) {
    const _Float16* xr = X + (size_t)b*C*512 + (size_t)t*512;
    const _Float16* rr = R + (size_t)b*C*512 + (size_t)t*512;
    float acc = 0.f;
    for (int l = 0; l < 512; ++l) acc += ((float)xr[l] + (float)rr[l]) * mrow[l];
    mp[t] = acc / msum_s;
  }
  __syncthreads();
  if (t < 16) {
    float v = gb1[t];
    for (int c = 0; c < C; ++c) v += gw1[t*C + c] * mp[c];
    v = v * (gg1[t] * BN_INV) + gbb1[t];
    h1s[t] = fmaxf(v, 0.f);
  }
  __syncthreads();
  if (t < C) {
    float v = gb2[t];
    #pragma unroll
    for (int j = 0; j < 16; ++j) v += gw2[t*16 + j] * h1s[j];
    xg[b*C + t] = v * (gg2[t] * BN_INV) + gbb2[t];
  }
}

// pass B: per-position 'l' branch + sigmoid gate -> O (B,C,L) f16
__global__ void k_affB(const _Float16* __restrict__ X, const _Float16* __restrict__ R,
                       const float* lw1, const float* lb1, const float* lg1, const float* lbb1,
                       const float* lw2, const float* lb2, const float* lg2, const float* lbb2,
                       const float* __restrict__ xg, _Float16* __restrict__ O, int C) {
  __shared__ _Float16 xt[128][64];
  __shared__ _Float16 rt[128][64];
  __shared__ float w1s[16*128];
  __shared__ float w2s[128*16];
  int b = blockIdx.x >> 3;
  int l0 = (blockIdx.x & 7) * 64;
  int t = threadIdx.x;  // 64 threads, one position each
  for (int i = t; i < 16*C; i += 64) w1s[i] = lw1[i];
  for (int i = t; i < C*16; i += 64) w2s[i] = lw2[i];
  for (int i = t; i < C*64; i += 64) {
    int c = i >> 6, p = i & 63;
    xt[c][p] = X[(size_t)b*C*512 + (size_t)c*512 + l0 + p];
    rt[c][p] = R[(size_t)b*C*512 + (size_t)c*512 + l0 + p];
  }
  __syncthreads();
  float h1[16];
  #pragma unroll
  for (int j = 0; j < 16; ++j) h1[j] = lb1[j];
  for (int c = 0; c < C; ++c) {
    float xa = (float)xt[c][t] + (float)rt[c][t];
    #pragma unroll
    for (int j = 0; j < 16; ++j) h1[j] += w1s[j*C + c] * xa;
  }
  #pragma unroll
  for (int j = 0; j < 16; ++j)
    h1[j] = fmaxf(h1[j] * (lg1[j] * BN_INV) + lbb1[j], 0.f);
  const float* xgb = xg + b*C;
  for (int c = 0; c < C; ++c) {
    float v = lb2[c];
    #pragma unroll
    for (int j = 0; j < 16; ++j) v += w2s[c*16 + j] * h1[j];
    v = v * (lg2[c] * BN_INV) + lbb2[c];
    float wei = 1.f / (1.f + __expf(-(v + xgb[c])));
    float xf = (float)xt[c][t], rf = (float)rt[c][t];
    O[(size_t)b*C*512 + (size_t)c*512 + l0 + t] = (_Float16)(2.f*xf*wei + 2.f*rf*(1.f - wei));
  }
}

// ---------------- attention head ----------------
__global__ void k_attn(const float* __restrict__ q, const _Float16* __restrict__ kmat,
                       const float* __restrict__ mask, float* __restrict__ pv) {
  __shared__ float probs[512];
  __shared__ float red[256];
  int b = blockIdx.x >> 3, hh = blockIdx.x & 7;
  int t = threadIdx.x;
  const float* qh = q + b*256 + hh*32;
  const float* mrow = mask + b*512;
  float sc[2];
  #pragma unroll
  for (int u = 0; u < 2; ++u) {
    int l = t + u*256;
    const _Float16* kr = kmat + ((size_t)(b*512 + l))*256 + hh*32;
    float d = 0.f;
    for (int i = 0; i < 32; ++i) d += qh[i] * (float)kr[i];
    d *= 0.17677669529663687f;  // 1/sqrt(32)
    if (!(mrow[l] > 0.f)) d = -1e9f;
    sc[u] = d; probs[l] = d;
  }
  __syncthreads();
  red[t] = fmaxf(probs[t], probs[t+256]); __syncthreads();
  for (int st = 128; st > 0; st >>= 1) { if (t < st) red[t] = fmaxf(red[t], red[t+st]); __syncthreads(); }
  float mx = red[0]; __syncthreads();
  float e0 = __expf(sc[0]-mx), e1 = __expf(sc[1]-mx);
  probs[t] = e0; probs[t+256] = e1;
  red[t] = e0 + e1; __syncthreads();
  for (int st = 128; st > 0; st >>= 1) { if (t < st) red[t] += red[t+st]; __syncthreads(); }
  float inv = 1.f / red[0]; __syncthreads();
  int d = t & 31, stripe = t >> 5;
  float acc = 0.f;
  for (int l = stripe*64; l < stripe*64 + 64; ++l)
    acc += probs[l] * (float)kmat[((size_t)(b*512 + l))*256 + hh*32 + d];
  red[t] = acc; __syncthreads();
  if (t < 32) {
    float s = 0.f;
    #pragma unroll
    for (int st = 0; st < 8; ++st) s += red[st*32 + t];
    pv[b*256 + hh*32 + t] = s * inv;
  }
}

__global__ void k_cat(const float* __restrict__ sr, const float* __restrict__ pv,
                      _Float16* __restrict__ cat) {
  int i = blockIdx.x * 256 + threadIdx.x;  // 256*512
  int b = i >> 9, c = i & 511;
  cat[i] = (_Float16)((c < 256) ? sr[b*256 + c] : pv[b*256 + (c-256)]);
}

__global__ void k_final(const _Float16* __restrict__ f, const float* __restrict__ pw2,
                        const float* __restrict__ pb2, float* __restrict__ out) {
  __shared__ float red[256];
  int b = blockIdx.x, t = threadIdx.x;
  red[t] = (float)f[b*256 + t] * pw2[t];
  __syncthreads();
  for (int st = 128; st > 0; st >>= 1) { if (t < st) red[t] += red[t+st]; __syncthreads(); }
  if (t == 0) out[b] = red[0] + pb2[0];
}

// ---------------- host orchestration ----------------
extern "C" void kernel_launch(void* const* d_in, const int* in_sizes, int n_in,
                              void* d_out, int out_size, void* d_ws, size_t ws_size,
                              hipStream_t stream) {
  (void)in_sizes; (void)n_in; (void)out_size; (void)ws_size;
  // JAX pytree (sorted-key) flattening of setup_inputs() dict:
  const int*   edge_attr  = (const int*)  d_in[0];
  const int*   edge_index = (const int*)  d_in[1];
  const float* mask       = (const float*)d_in[2];
  int i = 3;
  // params.aff1 (C=64): gb1,gb2,gbb1,gbb2,gg1,gg2,gw1,gw2,lb1,lb2,lbb1,lbb2,lg1,lg2,lw1,lw2
  const float *a1_gb1=(const float*)d_in[i++], *a1_gb2=(const float*)d_in[i++],
              *a1_gbb1=(const float*)d_in[i++], *a1_gbb2=(const float*)d_in[i++],
              *a1_gg1=(const float*)d_in[i++], *a1_gg2=(const float*)d_in[i++],
              *a1_gw1=(const float*)d_in[i++], *a1_gw2=(const float*)d_in[i++],
              *a1_lb1=(const float*)d_in[i++], *a1_lb2=(const float*)d_in[i++],
              *a1_lbb1=(const float*)d_in[i++], *a1_lbb2=(const float*)d_in[i++],
              *a1_lg1=(const float*)d_in[i++], *a1_lg2=(const float*)d_in[i++],
              *a1_lw1=(const float*)d_in[i++], *a1_lw2=(const float*)d_in[i++];
  // params.aff2 (C=128)
  const float *a2_gb1=(const float*)d_in[i++], *a2_gb2=(const float*)d_in[i++],
              *a2_gbb1=(const float*)d_in[i++], *a2_gbb2=(const float*)d_in[i++],
              *a2_gg1=(const float*)d_in[i++], *a2_gg2=(const float*)d_in[i++],
              *a2_gw1=(const float*)d_in[i++], *a2_gw2=(const float*)d_in[i++],
              *a2_lb1=(const float*)d_in[i++], *a2_lb2=(const float*)d_in[i++],
              *a2_lbb1=(const float*)d_in[i++], *a2_lbb2=(const float*)d_in[i++],
              *a2_lg1=(const float*)d_in[i++], *a2_lg2=(const float*)d_in[i++],
              *a2_lw1=(const float*)d_in[i++], *a2_lw2=(const float*)d_in[i++];
  const float *attb=(const float*)d_in[i++], *attw=(const float*)d_in[i++];
  const float *b1=(const float*)d_in[i++], *b2=(const float*)d_in[i++];
  const float *bnb=(const float*)d_in[i++], *bng=(const float*)d_in[i++];
  const float *cb1=(const float*)d_in[i++], *cb2=(const float*)d_in[i++], *cb3=(const float*)d_in[i++];
  const float *cbb1=(const float*)d_in[i++], *cbb2=(const float*)d_in[i++], *cbb3=(const float*)d_in[i++];
  const float *cg1=(const float*)d_in[i++], *cg2=(const float*)d_in[i++], *cg3=(const float*)d_in[i++];
  const float *cw1=(const float*)d_in[i++], *cw2=(const float*)d_in[i++], *cw3=(const float*)d_in[i++];
  const float *ee1=(const float*)d_in[i++], *ee2=(const float*)d_in[i++],
              *ee3=(const float*)d_in[i++], *ee4=(const float*)d_in[i++];
  const float *l1b=(const float*)d_in[i++], *l1w=(const float*)d_in[i++];
  const float *pb1=(const float*)d_in[i++], *pb2=(const float*)d_in[i++];
  const float *pw1=(const float*)d_in[i++], *pw2=(const float*)d_in[i++];
  const float *tb1=(const float*)d_in[i++], *tb2=(const float*)d_in[i++];
  const float *tw1=(const float*)d_in[i++], *tw2=(const float*)d_in[i++];
  const float *w1=(const float*)d_in[i++], *w2=(const float*)d_in[i++];
  const float *wemb=(const float*)d_in[i++];
  const float *xe1=(const float*)d_in[i++], *xe2=(const float*)d_in[i++], *xe3=(const float*)d_in[i++],
              *xe4=(const float*)d_in[i++], *xe5=(const float*)d_in[i++], *xe6=(const float*)d_in[i++],
              *xe7=(const float*)d_in[i++];
  const int *ptr=(const int*)d_in[i++], *words=(const int*)d_in[i++], *x=(const int*)d_in[i++];

  // ---- workspace bump allocator ----
  char* base = (char*)d_ws;
  size_t off = 0;
  auto alloc = [&](size_t bytes) -> void* {
    void* p = base + off; off = (off + bytes + 255) & ~(size_t)255; return p;
  };
  _Float16* w1h   = (_Float16*)alloc((size_t)5*256*512*2);
  _Float16* w2h   = (_Float16*)alloc((size_t)5*512*256*2);
  _Float16* cw1T  = (_Float16*)alloc((size_t)15*128*256*2);
  _Float16* cw2T  = (_Float16*)alloc((size_t)15*64*128*2);
  _Float16* cw3T  = (_Float16*)alloc((size_t)15*32*64*2);
  _Float16* tw1T  = (_Float16*)alloc((size_t)15*64*32*2);
  _Float16* tw2T  = (_Float16*)alloc((size_t)15*128*64*2);
  _Float16* l1wh  = (_Float16*)alloc((size_t)128*256*2);
  _Float16* attwh = (_Float16*)alloc((size_t)256*256*2);
  _Float16* pw1h  = (_Float16*)alloc((size_t)512*256*2);
  _Float16* wembh = (_Float16*)alloc((size_t)10000*256*2);
  float* super_rep= (float*)alloc((size_t)256*256*4);
  float* qbuf     = (float*)alloc((size_t)256*256*4);
  float* pvb      = (float*)alloc((size_t)256*256*4);
  float* xg1      = (float*)alloc((size_t)256*64*4);
  float* xg2      = (float*)alloc((size_t)256*128*4);
  _Float16* catb  = (_Float16*)alloc((size_t)256*512*2);
  _Float16* fc1   = (_Float16*)alloc((size_t)256*256*2);
  size_t big = off;
  // phase 1 (GNN)
  float*    h   = (float*)(base + big);
  float*    agg = h + (size_t)100000*256;
  _Float16* hm  = (_Float16*)(agg + (size_t)100000*256);
  // phase 2 (conv) — reuses the same region (GNN results consumed via super_rep first)
  char* p2 = base + big;
  _Float16* hp  = (_Float16*)p2; p2 += (size_t)256*256*512*2;
  _Float16* h1b = (_Float16*)p2; p2 += (size_t)256*128*512*2;
  _Float16* h2b = (_Float16*)p2; p2 += (size_t)256*64*512*2;
  _Float16* h3b = (_Float16*)p2; p2 += (size_t)256*32*512*2;
  _Float16* t1b = (_Float16*)p2; p2 += (size_t)256*64*512*2;
  _Float16* a1b = (_Float16*)p2; p2 += (size_t)256*64*512*2;
  _Float16* t2b = (_Float16*)p2; p2 += (size_t)256*128*512*2;
  _Float16* a2b = (_Float16*)p2; p2 += (size_t)256*128*512*2;
  _Float16* a2t = (_Float16*)p2; p2 += (size_t)131072*128*2;
  _Float16* prot= (_Float16*)p2; p2 += (size_t)131072*256*2;

  auto g1 = [](long long n){ return (unsigned)((n + 255) / 256); };

  // ---- weight conversions (f32 -> f16, (K,N) layouts already match x@W) ----
  k_cvt_f16<<<g1(5*256*512), 256, 0, stream>>>(w1, w1h, 5*256*512);
  k_cvt_f16<<<g1(5*512*256), 256, 0, stream>>>(w2, w2h, 5*512*256);
  k_cvt_f16<<<g1(128*256), 256, 0, stream>>>(l1w, l1wh, 128*256);
  k_cvt_f16<<<g1(256*256), 256, 0, stream>>>(attw, attwh, 256*256);
  k_cvt_f16<<<g1(512*256), 256, 0, stream>>>(pw1, pw1h, 512*256);
  k_cvt_f16<<<g1(10000*256), 256, 0, stream>>>(wemb, wembh, 10000*256);
  k_convw<<<g1(15*128*256), 256, 0, stream>>>(cw1, cw1T, 128, 256, 0);
  k_convw<<<g1(15*64*128), 256, 0, stream>>>(cw2, cw2T, 64, 128, 0);
  k_convw<<<g1(15*32*64), 256, 0, stream>>>(cw3, cw3T, 32, 64, 0);
  k_convw<<<g1(15*64*32), 256, 0, stream>>>(tw1, tw1T, 64, 32, 1);
  k_convw<<<g1(15*128*64), 256, 0, stream>>>(tw2, tw2T, 128, 64, 1);

  // ---- GNN ----
  k_node_embed<<<100000, 256, 0, stream>>>(x, xe1, xe2, xe3, xe4, xe5, xe6, xe7, h);
  for (int l = 0; l < 5; ++l) {
    const float *e1 = ee1 + (size_t)l*7*256, *e2 = ee2 + (size_t)l*3*256,
                *e3 = ee3 + (size_t)l*3*256, *e4 = ee4 + (size_t)l*6*256;
    k_init_agg<<<100000, 256, 0, stream>>>(h, agg, e1, e2, e3, e4);
    k_edge_scatter<<<400000, 256, 0, stream>>>(edge_index, edge_attr, h, agg, e1, e2, e3, e4);
    // hm = relu(agg @ w1 + b1), f16 out
    k_gemm<false><<<dim3(512/64, (100000+127)/128), 256, 0, stream>>>(
        agg, w1h + (size_t)l*256*512, hm, b1 + l*512, nullptr, nullptr, 100000, 512, 256, 1|8);
    // h = bn(hm @ w2 + b2) [, elu], f32 out
    k_gemm<true><<<dim3(256/64, (100000+127)/128), 256, 0, stream>>>(
        hm, w2h + (size_t)l*512*256, h, b2 + l*256, bng + l*256, bnb + l*256,
        100000, 256, 512, 4 | ((l < 4) ? 2 : 0));
  }
  k_super<<<256, 256, 0, stream>>>(h, ptr, super_rep);

  // ---- conv stack (word path) ----
  k_hp_gather<<<g1((long long)256*256*512), 256, 0, stream>>>(words, wembh, hp);
  k_conv_wmma<<<dim3(8, 8, 256), 128, 0, stream>>>(hp,  cw1T, h1b, cb1, cg1, cbb1, 256, 128, 1|4);
  k_conv_wmma<<<dim3(8, 4, 256), 128, 0, stream>>>(h1b, cw2T, h2b, cb2, cg2, cbb2, 128, 64, 1|4);
  k_conv_wmma<<<dim3(8, 2, 256), 128, 0, stream>>>(h2b, cw3T, h3b, cb3, cg3, cbb3, 64, 32, 4);
  k_conv_wmma<<<dim3(8, 4, 256), 128, 0, stream>>>(h3b, tw1T, t1b, tb1, nullptr, nullptr, 32, 64, 0);
  k_affA<<<256, 128, 0, stream>>>(t1b, h2b, mask, a1_gw1, a1_gb1, a1_gg1, a1_gbb1,
                                  a1_gw2, a1_gb2, a1_gg2, a1_gbb2, xg1, 64);
  k_affB<<<256*8, 64, 0, stream>>>(t1b, h2b, a1_lw1, a1_lb1, a1_lg1, a1_lbb1,
                                   a1_lw2, a1_lb2, a1_lg2, a1_lbb2, xg1, a1b, 64);
  k_conv_wmma<<<dim3(8, 8, 256), 128, 0, stream>>>(a1b, tw2T, t2b, tb2, nullptr, nullptr, 64, 128, 0);
  k_affA<<<256, 128, 0, stream>>>(t2b, h1b, mask, a2_gw1, a2_gb1, a2_gg1, a2_gbb1,
                                  a2_gw2, a2_gb2, a2_gg2, a2_gbb2, xg2, 128);
  k_affB<<<256*8, 64, 0, stream>>>(t2b, h1b, a2_lw1, a2_lb1, a2_lg1, a2_lbb1,
                                   a2_lw2, a2_lb2, a2_lg2, a2_lbb2, xg2, a2b, 128);
  k_a2t<<<g1((long long)131072*128), 256, 0, stream>>>(a2b, a2t);
  // prot = a2^T @ l1w + l1b  (M=131072, K=128, N=256), f16 out
  k_gemm<true><<<dim3(4, 1024), 256, 0, stream>>>(a2t, l1wh, prot, l1b, nullptr, nullptr,
                                                  131072, 256, 128, 8);
  // q = super_rep @ attw + attb, f32 out
  k_gemm<false><<<dim3(4, 2), 256, 0, stream>>>(super_rep, attwh, qbuf, attb, nullptr, nullptr,
                                                256, 256, 256, 0);
  k_attn<<<2048, 256, 0, stream>>>(qbuf, prot, mask, pvb);
  k_cat<<<512, 256, 0, stream>>>(super_rep, pvb, catb);
  // fc1 = elu(cat @ pw1 + pb1), f16 out
  k_gemm<true><<<dim3(4, 2), 256, 0, stream>>>(catb, pw1h, fc1, pb1, nullptr, nullptr,
                                               256, 256, 512, 2|8);
  k_final<<<256, 256, 0, stream>>>(fc1, pw2, pb2, (float*)d_out);
}
